// LivingBlock_29678224016188
// MI455X (gfx1250) — compile-verified, matrix operand order checked by reading, not run on hardware
//
#include <hip/hip_runtime.h>
#include <math.h>

// Problem constants (match reference)
#define BB  2
#define TT  512
#define DDIM 512
#define HH  8
#define HDIM 64
#define FFD 2048
#define MMROWS (BB*TT)        // 1024
#define DSQ (DDIM*DDIM)       // 262144

typedef __attribute__((ext_vector_type(16))) _Float16 v16h;
typedef __attribute__((ext_vector_type(8)))  float    v8f;
typedef __attribute__((ext_vector_type(4)))  float    v4f;

// ---------------------------------------------------------------------------
// WMMA GEMM: C[M,N] = act(A[M,K] @ W[N,K]^T + bias) + resid
// grid = (N/64, M/64), block = 128 (4 waves). Each wave owns a 64x16 C strip:
// 4 fp32 accumulators sharing one B fragment -> 4 v_wmma per K-chunk.
// Per the CDNA5 16-bit fragment layout each lane needs two contiguous
// 8-float K-runs (k0+8*hsel, k0+16+8*hsel): 4x global_load_b128 per matrix,
// f32->f16 in registers. Fragments are double-buffered across K-chunks so
// the next chunk's 20 b128 loads issue while the current 4 WMMAs execute
// (no loadcnt-0 stalls, no WMMA->VALU WAR hazard NOPs).
// ---------------------------------------------------------------------------
struct FragSet { v16h a0, a1, a2, a3, b; };

__device__ __forceinline__ v16h cvt_frag(const float* __restrict__ p, int ka, int kb) {
  const v4f x0 = *(const v4f*)(p + ka);
  const v4f x1 = *(const v4f*)(p + ka + 4);
  const v4f x2 = *(const v4f*)(p + kb);
  const v4f x3 = *(const v4f*)(p + kb + 4);
  v16h f;
#pragma unroll
  for (int e = 0; e < 4; ++e) {
    f[e]      = (_Float16)x0[e];
    f[4 + e]  = (_Float16)x1[e];
    f[8 + e]  = (_Float16)x2[e];
    f[12 + e] = (_Float16)x3[e];
  }
  return f;
}

__device__ __forceinline__ FragSet load_frags(const float* __restrict__ ar0,
                                              const float* __restrict__ wrow,
                                              size_t mstride, int k0, int hsel) {
  FragSet f;
  const int ka = k0 + 8 * hsel;
  const int kb = k0 + 16 + 8 * hsel;
  f.b  = cvt_frag(wrow, ka, kb);
  f.a0 = cvt_frag(ar0, ka, kb);
  f.a1 = cvt_frag(ar0 + mstride, ka, kb);
  f.a2 = cvt_frag(ar0 + 2 * mstride, ka, kb);
  f.a3 = cvt_frag(ar0 + 3 * mstride, ka, kb);
  return f;
}

__global__ __launch_bounds__(128)
void k_gemm_wmma(const float* __restrict__ A, const float* __restrict__ W,
                 const float* __restrict__ bias, const float* __restrict__ resid,
                 float* __restrict__ C, int N, int K, int act) {
  const int tid  = threadIdx.x;
  const int wv   = tid >> 5;           // wave id 0..3 (N sub-tile)
  const int lane = tid & 31;
  const int row  = lane & 15;          // M index within tile (A), N index (B/C)
  const int hsel = lane >> 4;          // lane-half select
  const int m0   = blockIdx.y * 64;
  const int n0   = blockIdx.x * 64 + wv * 16;

  const float* wrow = W + (size_t)(n0 + row) * K;
  const float* ar0  = A + (size_t)(m0 + row) * K;
  const size_t mstride = (size_t)16 * K;

  v8f acc[4] = {{}, {}, {}, {}};
  FragSet cur = load_frags(ar0, wrow, mstride, 0, hsel);
  for (int k0 = 32; k0 < K; k0 += 32) {
    FragSet nxt = load_frags(ar0, wrow, mstride, k0, hsel);  // in flight over WMMAs
    acc[0] = __builtin_amdgcn_wmma_f32_16x16x32_f16(false, cur.a0, false, cur.b,
                                                    (short)0, acc[0], false, false);
    acc[1] = __builtin_amdgcn_wmma_f32_16x16x32_f16(false, cur.a1, false, cur.b,
                                                    (short)0, acc[1], false, false);
    acc[2] = __builtin_amdgcn_wmma_f32_16x16x32_f16(false, cur.a2, false, cur.b,
                                                    (short)0, acc[2], false, false);
    acc[3] = __builtin_amdgcn_wmma_f32_16x16x32_f16(false, cur.a3, false, cur.b,
                                                    (short)0, acc[3], false, false);
    cur = nxt;
  }
  acc[0] = __builtin_amdgcn_wmma_f32_16x16x32_f16(false, cur.a0, false, cur.b,
                                                  (short)0, acc[0], false, false);
  acc[1] = __builtin_amdgcn_wmma_f32_16x16x32_f16(false, cur.a1, false, cur.b,
                                                  (short)0, acc[1], false, false);
  acc[2] = __builtin_amdgcn_wmma_f32_16x16x32_f16(false, cur.a2, false, cur.b,
                                                  (short)0, acc[2], false, false);
  acc[3] = __builtin_amdgcn_wmma_f32_16x16x32_f16(false, cur.a3, false, cur.b,
                                                  (short)0, acc[3], false, false);

  // C/D layout: VGPR e -> M = e + 8*hsel, N = lane&15
#pragma unroll
  for (int mt = 0; mt < 4; ++mt) {
#pragma unroll
    for (int e = 0; e < 8; ++e) {
      const int gm = m0 + mt * 16 + e + 8 * hsel;
      const int gn = n0 + row;
      float vv = acc[mt][e];
      if (bias)  vv += bias[gn];
      if (act == 1) vv = 0.5f * vv * (1.0f + erff(vv * 0.70710678118654752f)); // exact GELU
      if (resid) vv += resid[(size_t)gm * N + gn];
      C[(size_t)gm * N + gn] = vv;
    }
  }
}

// ---------------------------------------------------------------------------
// LayerNorm over D=512; one block (256 thr) per row, two elems per thread.
// ---------------------------------------------------------------------------
__global__ __launch_bounds__(256)
void k_layernorm(const float* __restrict__ src, const float* __restrict__ g,
                 const float* __restrict__ b, float* __restrict__ dst) {
  __shared__ float red[256];
  const int r = blockIdx.x, tid = threadIdx.x;
  const float x0 = src[(size_t)r * DDIM + tid];
  const float x1 = src[(size_t)r * DDIM + tid + 256];
  red[tid] = x0 + x1; __syncthreads();
  for (int s = 128; s > 0; s >>= 1) { if (tid < s) red[tid] += red[tid + s]; __syncthreads(); }
  const float mean = red[0] * (1.0f / DDIM);
  __syncthreads();
  const float d0 = x0 - mean, d1 = x1 - mean;
  red[tid] = d0 * d0 + d1 * d1; __syncthreads();
  for (int s = 128; s > 0; s >>= 1) { if (tid < s) red[tid] += red[tid + s]; __syncthreads(); }
  const float rstd = rsqrtf(red[0] * (1.0f / DDIM) + 1e-5f);
  dst[(size_t)r * DDIM + tid]       = d0 * rstd * g[tid] + b[tid];
  dst[(size_t)r * DDIM + tid + 256] = d1 * rstd * g[tid + 256] + b[tid + 256];
}

// ---------------------------------------------------------------------------
// Differential causal attention core: one block per (b,h,query row).
// ---------------------------------------------------------------------------
__global__ __launch_bounds__(128)
void k_diff_attn(const float* __restrict__ q, const float* __restrict__ k1,
                 const float* __restrict__ k2, const float* __restrict__ v,
                 const float* __restrict__ lambda_logit, float* __restrict__ o) {
  __shared__ __align__(16) float sq[HDIM];
  __shared__ float sc1[TT];
  __shared__ float sc2[TT];
  __shared__ float red[128];
  const int qi = blockIdx.x, h = blockIdx.y, b = blockIdx.z;
  const int tid = threadIdx.x;
  const size_t base = (size_t)(b * TT) * DDIM + h * HDIM;
  if (tid < HDIM) sq[tid] = q[base + (size_t)qi * DDIM + tid];
  __syncthreads();

  const float scale = 0.125f;  // HD^-0.5
  float mx1 = -3.0e38f, mx2 = -3.0e38f;
  for (int k = tid; k <= qi; k += 128) {
    const float* kp1 = &k1[base + (size_t)k * DDIM];
    const float* kp2 = &k2[base + (size_t)k * DDIM];
    float a1 = 0.f, a2 = 0.f;
#pragma unroll
    for (int d = 0; d < HDIM; d += 4) {
      const v4f qv  = *(const v4f*)(sq + d);
      const v4f kv1 = *(const v4f*)(kp1 + d);
      const v4f kv2 = *(const v4f*)(kp2 + d);
      a1 += qv[0]*kv1[0] + qv[1]*kv1[1] + qv[2]*kv1[2] + qv[3]*kv1[3];
      a2 += qv[0]*kv2[0] + qv[1]*kv2[1] + qv[2]*kv2[2] + qv[3]*kv2[3];
    }
    a1 *= scale; a2 *= scale;
    sc1[k] = a1; sc2[k] = a2;
    mx1 = fmaxf(mx1, a1); mx2 = fmaxf(mx2, a2);
  }
  red[tid] = mx1; __syncthreads();
  for (int s = 64; s > 0; s >>= 1) { if (tid < s) red[tid] = fmaxf(red[tid], red[tid + s]); __syncthreads(); }
  mx1 = red[0]; __syncthreads();
  red[tid] = mx2; __syncthreads();
  for (int s = 64; s > 0; s >>= 1) { if (tid < s) red[tid] = fmaxf(red[tid], red[tid + s]); __syncthreads(); }
  mx2 = red[0]; __syncthreads();

  float s1 = 0.f, s2 = 0.f;
  for (int k = tid; k <= qi; k += 128) {
    const float e1 = __expf(sc1[k] - mx1);
    const float e2 = __expf(sc2[k] - mx2);
    sc1[k] = e1; sc2[k] = e2; s1 += e1; s2 += e2;
  }
  red[tid] = s1; __syncthreads();
  for (int s = 64; s > 0; s >>= 1) { if (tid < s) red[tid] += red[tid + s]; __syncthreads(); }
  s1 = red[0]; __syncthreads();
  red[tid] = s2; __syncthreads();
  for (int s = 64; s > 0; s >>= 1) { if (tid < s) red[tid] += red[tid + s]; __syncthreads(); }
  s2 = red[0]; __syncthreads();

  const float lam = 1.0f / (1.0f + __expf(-lambda_logit[h]));
  float l1 = 0.f;
  for (int k = tid; k <= qi; k += 128) {
    const float a = sc1[k] / s1 - lam * (sc2[k] / s2);
    sc1[k] = a; l1 += fabsf(a);
  }
  red[tid] = l1; __syncthreads();
  for (int s = 64; s > 0; s >>= 1) { if (tid < s) red[tid] += red[tid + s]; __syncthreads(); }
  const float denom = fmaxf(red[0], 1e-6f);
  __syncthreads();

  if (tid < HDIM) {
    float accv = 0.f;
    for (int k = 0; k <= qi; ++k) accv += sc1[k] * v[base + (size_t)k * DDIM + tid];
    o[base + (size_t)qi * DDIM + tid] = accv / denom;
  }
}

// ---------------------------------------------------------------------------
// TTT scan. Fast weights factored as fw = c * F (c scalar per batch) so the
// per-step Frobenius norm clip is a scalar update, not an extra matrix pass.
// SCAL[4b+0]=c, +1=lrm, +2=eta(=inner_lr*lrm), +3=sumsq(F)
// ---------------------------------------------------------------------------
__global__ void k_ttt_init_fw(const float* __restrict__ bw, float* __restrict__ F,
                              float* __restrict__ mom) {
  const int i = blockIdx.x * 256 + threadIdx.x;   // over BB*DSQ
  F[i] = bw[i & (DSQ - 1)];
  mom[i] = 0.f;
}

__global__ void k_ttt_init_small(float* __restrict__ ema, float* __restrict__ scal) {
  const int i = blockIdx.x * 256 + threadIdx.x;
  if (i < BB * DDIM) ema[i] = 0.f;
  if (i < BB) { scal[4*i+0] = 1.f; scal[4*i+1] = 1.f; scal[4*i+2] = 0.f; scal[4*i+3] = 0.f; }
}

__global__ __launch_bounds__(128)
void k_ttt_matvec(const float* __restrict__ F, const float* __restrict__ hn,
                  const float* __restrict__ bias, const float* __restrict__ scal,
                  float* __restrict__ ys, float* __restrict__ err, int t) {
  __shared__ __align__(16) float xs[DDIM];
  const int b = blockIdx.y, tid = threadIdx.x;
  const float* xt = &hn[(size_t)(b * TT + t) * DDIM];
  for (int i = tid; i < DDIM; i += 128) xs[i] = xt[i];
  __syncthreads();
  const int row = blockIdx.x * 128 + tid;
  const float* fr = &F[(size_t)b * DSQ + (size_t)row * DDIM];
  float acc = 0.f;
#pragma unroll 4
  for (int j = 0; j < DDIM; j += 4) {
    const v4f fv = *(const v4f*)(fr + j);
    const v4f xv = *(const v4f*)(xs + j);
    acc += fv[0]*xv[0] + fv[1]*xv[1] + fv[2]*xv[2] + fv[3]*xv[3];
  }
  const float c = scal[4 * b + 0];
  const float y = c * acc + bias[row];
  ys[(size_t)(b * TT + t) * DDIM + row] = y;
  err[b * DDIM + row] = (y - xs[row]) * (1.0f / DDIM);
}

__global__ __launch_bounds__(256)
void k_ttt_scalars(const float* __restrict__ hn, float* __restrict__ ema,
                   float* __restrict__ scal, const float* __restrict__ log_inner_lr,
                   int t) {
  __shared__ float red[256];
  const int b = blockIdx.x, tid = threadIdx.x;
  const float* xt = &hn[(size_t)(b * TT + t) * DDIM];
  const float x0 = xt[tid], x1 = xt[tid + 256];
  const float e0 = ema[b * DDIM + tid], e1 = ema[b * DDIM + tid + 256];
  red[tid] = fabsf(x0 - e0) + fabsf(x1 - e1);
  __syncthreads();
  for (int s = 128; s > 0; s >>= 1) { if (tid < s) red[tid] += red[tid + s]; __syncthreads(); }
  if (tid == 0) {
    const float nov = red[0] * (1.0f / DDIM);
    const float lrs = fminf(fmaxf(1.0f + 3.0f * nov, 0.5f), 3.0f);
    const float lrm = 0.95f * scal[4 * b + 1] + 0.05f * lrs;
    scal[4 * b + 1] = lrm;
    const float il = fminf(fmaxf(__expf(log_inner_lr[0]), 1e-5f), 1.0f);
    scal[4 * b + 2] = il * lrm;
    scal[4 * b + 3] = 0.f;   // reset sumsq accumulator for this step
  }
  ema[b * DDIM + tid]       = 0.95f * e0 + 0.05f * x0;
  ema[b * DDIM + tid + 256] = 0.95f * e1 + 0.05f * x1;
}

__global__ __launch_bounds__(256)
void k_ttt_update(float* __restrict__ F, float* __restrict__ mom,
                  const float* __restrict__ err, const float* __restrict__ hn,
                  float* __restrict__ scal, int t) {
  __shared__ float red[256];
  const int b = blockIdx.y, tid = threadIdx.x;
  const int idx = blockIdx.x * 256 + tid;           // 0..DSQ-1
  const int i = idx >> 9, j = idx & 511;
  const size_t p = (size_t)b * DSQ + idx;
  const float g  = err[b * DDIM + i] * hn[(size_t)(b * TT + t) * DDIM + j];
  const float m2 = 0.9f * mom[p] + 0.1f * g;
  mom[p] = m2;
  const float c   = scal[4 * b + 0];
  const float eta = scal[4 * b + 2];
  const float f2  = F[p] - (eta / c) * m2;          // fw2 = c*(F - (eta/c)*mom2)
  F[p] = f2;
  red[tid] = f2 * f2; __syncthreads();
  for (int s = 128; s > 0; s >>= 1) { if (tid < s) red[tid] += red[tid + s]; __syncthreads(); }
  if (tid == 0) atomicAdd(&scal[4 * b + 3], red[0]);
}

__global__ void k_ttt_cupdate(float* __restrict__ scal) {
  const int b = threadIdx.x;
  if (b < BB) {
    const float c    = scal[4 * b + 0];
    const float norm = fmaxf(c * sqrtf(scal[4 * b + 3]), 1e-6f);
    scal[4 * b + 0]  = c * fminf(32.0f / norm, 1.0f);
  }
}

__global__ void k_add(const float* __restrict__ a, const float* __restrict__ b,
                      float* __restrict__ c, int n) {
  const int i = blockIdx.x * 256 + threadIdx.x;
  if (i < n) c[i] = a[i] + b[i];
}

// ---------------------------------------------------------------------------
extern "C" void kernel_launch(void* const* d_in, const int* in_sizes, int n_in,
                              void* d_out, int out_size, void* d_ws, size_t ws_size,
                              hipStream_t stream) {
  const float* x    = (const float*)d_in[0];
  const float* Wq   = (const float*)d_in[1];
  const float* Wk1  = (const float*)d_in[2];
  const float* Wk2  = (const float*)d_in[3];
  const float* Wv   = (const float*)d_in[4];
  const float* Wo   = (const float*)d_in[5];
  const float* lamb = (const float*)d_in[6];
  const float* ln1g = (const float*)d_in[7];
  const float* ln1b = (const float*)d_in[8];
  const float* bw   = (const float*)d_in[9];
  const float* tbias= (const float*)d_in[10];
  const float* lil  = (const float*)d_in[11];
  const float* ln2g = (const float*)d_in[12];
  const float* ln2b = (const float*)d_in[13];
  const float* W1   = (const float*)d_in[14];
  const float* b1   = (const float*)d_in[15];
  const float* W2   = (const float*)d_in[16];
  const float* b2   = (const float*)d_in[17];
  const float* ln3g = (const float*)d_in[18];
  const float* ln3b = (const float*)d_in[19];
  float* out = (float*)d_out;
  float* ws  = (float*)d_ws;

  const size_t S = (size_t)MMROWS * DDIM;  // 524288 floats
  float* R0 = ws;             // xn -> attn o -> h2
  float* R1 = ws + 1 * S;     // q  -> hn -> z
  float* R2 = ws + 2 * S;     // k1 -> F
  float* R3 = ws + 3 * S;     // k2 -> mom
  float* R4 = ws + 4 * S;     // v  -> ys
  float* R5 = ws + 5 * S;     // h1
  float* R6 = ws + 6 * S;     // zf [MM, FF]
  float* ERR  = ws + 6 * S + (size_t)MMROWS * FFD;
  float* EMA  = ERR + BB * DDIM;
  float* SCAL = EMA + BB * DDIM;
  // total ws use: 6*S + MM*FF + 2*B*D + 16  ~= 5.25M floats (~21 MB)

  const dim3 bLN(256), gLN(MMROWS);
  const dim3 bG(128);
  const dim3 gP(DDIM / 64, MMROWS / 64);

  // --- sublayer 1: differential attention ---
  k_layernorm<<<gLN, bLN, 0, stream>>>(x, ln1g, ln1b, R0);
  k_gemm_wmma<<<gP, bG, 0, stream>>>(R0, Wq,  nullptr, nullptr, R1, DDIM, DDIM, 0);
  k_gemm_wmma<<<gP, bG, 0, stream>>>(R0, Wk1, nullptr, nullptr, R2, DDIM, DDIM, 0);
  k_gemm_wmma<<<gP, bG, 0, stream>>>(R0, Wk2, nullptr, nullptr, R3, DDIM, DDIM, 0);
  k_gemm_wmma<<<gP, bG, 0, stream>>>(R0, Wv,  nullptr, nullptr, R4, DDIM, DDIM, 0);
  k_diff_attn<<<dim3(TT, HH, BB), dim3(128), 0, stream>>>(R1, R2, R3, R4, lamb, R0);
  k_gemm_wmma<<<gP, bG, 0, stream>>>(R0, Wo, nullptr, x, R5, DDIM, DDIM, 0);  // h1

  // --- sublayer 2: TTT scan ---
  k_layernorm<<<gLN, bLN, 0, stream>>>(R5, ln2g, ln2b, R1);                   // hn
  float* Fw = R2; float* Mo = R3; float* Ys = R4;
  k_ttt_init_fw<<<dim3((BB * DSQ) / 256), dim3(256), 0, stream>>>(bw, Fw, Mo);
  k_ttt_init_small<<<dim3(4), dim3(256), 0, stream>>>(EMA, SCAL);
  for (int t = 0; t < TT; ++t) {
    k_ttt_matvec <<<dim3(4, BB),         dim3(128), 0, stream>>>(Fw, R1, tbias, SCAL, Ys, ERR, t);
    k_ttt_scalars<<<dim3(BB),            dim3(256), 0, stream>>>(R1, EMA, SCAL, lil, t);
    k_ttt_update <<<dim3(DSQ / 256, BB), dim3(256), 0, stream>>>(Fw, Mo, ERR, R1, SCAL, t);
    k_ttt_cupdate<<<dim3(1),             dim3(32),  0, stream>>>(SCAL);
  }
  k_add<<<dim3((int)(S / 256)), dim3(256), 0, stream>>>(R5, Ys, R0, (int)S);  // h2

  // --- sublayer 3: MLP ---
  k_layernorm<<<gLN, bLN, 0, stream>>>(R0, ln3g, ln3b, R1);                   // z
  k_gemm_wmma<<<dim3(FFD / 64, MMROWS / 64), bG, 0, stream>>>(R1, W1, b1, nullptr, R6, FFD, DDIM, 1);
  k_gemm_wmma<<<dim3(DDIM / 64, MMROWS / 64), bG, 0, stream>>>(R6, W2, b2, R0, out, DDIM, FFD, 0);
}
